// Decoder_50448685858992
// MI455X (gfx1250) — matrix operand order repaired and checked by reading, not compiled, not run on hardware
//
#include <hip/hip_runtime.h>

// Decoder layer for MI455X (gfx1250, wave32): all GEMM-like work on
// v_wmma_f32_16x16x32_bf16 (f32 accumulate), flash-attention with online
// softmax, explicit s_wait_dscnt for intra-wave LDS staging of P/V tiles.

constexpr int SEQ   = 2048;
constexpr int DMODEL = 1024;
constexpr int NHEAD = 16;
constexpr int HDIM  = 64;
constexpr int FFDIM = 4096;

typedef __attribute__((ext_vector_type(16))) __bf16 v16bf;
typedef __attribute__((ext_vector_type(8)))  __bf16 v8bf;
typedef __attribute__((ext_vector_type(8)))  float  v8f;

__device__ __forceinline__ void wait_lds() {
  asm volatile("s_wait_dscnt 0" ::: "memory");
}

__device__ __forceinline__ v8f wmma_bf16(v16bf a, v16bf b, v8f c) {
  // (neg_a, A, neg_b, B, c_mod, C, reuse_a, reuse_b)
  return __builtin_amdgcn_wmma_f32_16x16x32_bf16(false, a, false, b, (short)0, c,
                                                 false, false);
}

// ---------------------------------------------------------------------------
// Generic tiled GEMM: C[M][N] = A[M][K] @ B[K][N] (+bias) (+relu)
// A,B converted to bf16 through LDS. 128 threads = 4 waves, 64x64 block tile,
// each wave a 32x32 sub-tile (2x2 WMMA fragments), BK = 32 (one WMMA K step).
// blockIdx.z batches B / bias / C (per-head projections).
// ---------------------------------------------------------------------------
template<typename TA, typename TB, bool BIAS, bool RELU, bool OUT_BF16>
__global__ __launch_bounds__(128)
void gemm_ws_kernel(const TA* __restrict__ A, const TB* __restrict__ B,
                    const float* __restrict__ bias, void* __restrict__ Cout,
                    int M, int N, int K, long sB, long sBias, long sC)
{
  __shared__ __bf16 As[64][32];   // [m][k]
  __shared__ __bf16 Bs[64][32];   // transposed: [n][k]

  const int tid  = threadIdx.x;
  const int lane = tid & 31;
  const int wave = tid >> 5;
  const int wm = wave >> 1, wn = wave & 1;
  const int hi = lane >> 4, l16 = lane & 15;

  const int  m0 = blockIdx.x * 64;
  const int  n0 = blockIdx.y * 64;
  const long z  = blockIdx.z;
  const TB* Bz = B + z * sB;

  const int ar = tid >> 1, ac0 = (tid & 1) * 16;   // A: 16 elems/thread
  const int bk = tid >> 2, bn0 = (tid & 3) * 16;   // B: 16 elems/thread

  v8f c[2][2] = {};

  for (int k0 = 0; k0 < K; k0 += 32) {
    {
      const TA* Ap = A + (size_t)(m0 + ar) * K + (k0 + ac0);
      #pragma unroll
      for (int i = 0; i < 16; ++i) As[ar][ac0 + i] = (__bf16)(float)Ap[i];
      const TB* Bp = Bz + (size_t)(k0 + bk) * N + (n0 + bn0);
      #pragma unroll
      for (int i = 0; i < 16; ++i) Bs[bn0 + i][bk] = (__bf16)(float)Bp[i];
    }
    __syncthreads();

    v16bf a[2], b[2];
    #pragma unroll
    for (int mt = 0; mt < 2; ++mt) {
      // 16-bit A layout: lanes 0-15 K 0-7 / 16-23, lanes 16-31 K 8-15 / 24-31
      const __bf16* p = &As[wm * 32 + mt * 16 + l16][hi * 8];
      #pragma unroll
      for (int e = 0; e < 8; ++e) { a[mt][e] = p[e]; a[mt][e + 8] = p[16 + e]; }
    }
    #pragma unroll
    for (int nt = 0; nt < 2; ++nt) {
      // 16-bit B layout: lanes 0-15 K 0-7 / 16-23, lanes 16-31 K 8-15 / 24-31, col = lane%16
      const __bf16* p = &Bs[wn * 32 + nt * 16 + l16][hi * 8];
      #pragma unroll
      for (int e = 0; e < 8; ++e) { b[nt][e] = p[e]; b[nt][e + 8] = p[16 + e]; }
    }
    #pragma unroll
    for (int mt = 0; mt < 2; ++mt)
      #pragma unroll
      for (int nt = 0; nt < 2; ++nt)
        c[mt][nt] = wmma_bf16(a[mt], b[nt], c[mt][nt]);
    __syncthreads();
  }

  #pragma unroll
  for (int mt = 0; mt < 2; ++mt)
    #pragma unroll
    for (int nt = 0; nt < 2; ++nt)
      #pragma unroll
      for (int r = 0; r < 8; ++r) {
        const int gm = m0 + wm * 32 + mt * 16 + hi * 8 + r;  // C: M = r + 8*hi
        const int gn = n0 + wn * 32 + nt * 16 + l16;         // C: N = lane%16
        float v = c[mt][nt][r];
        if (BIAS) v += bias[z * sBias + gn];
        if (RELU) v = fmaxf(v, 0.0f);
        const size_t idx = (size_t)(z * sC) + (size_t)gm * N + gn;
        if (OUT_BF16) ((__bf16*)Cout)[idx] = (__bf16)v;
        else          ((float*)Cout)[idx]  = v;
      }
}

// ---------------------------------------------------------------------------
// Flash attention. Q/K/V: bf16 [H][S][64]. Out: f32 [S][D] at column h*64.
// 256 threads = 8 waves; each wave owns a 16-row query tile, iterates over
// keys in 32-wide tiles with online softmax kept in registers. P and V tiles
// are staged per-wave through LDS so all WMMA fragments come from contiguous
// ds_load_b128 reads (no strided scalar global loads).
// ---------------------------------------------------------------------------
template<bool CAUSAL>
__global__ __launch_bounds__(256)
void attn_kernel(const __bf16* __restrict__ Q, const __bf16* __restrict__ Km,
                 const __bf16* __restrict__ Vm, float* __restrict__ Out)
{
  __shared__ __bf16 plds[8][16][32];   // per-wave P tile (rows x keys)
  __shared__ __bf16 Vs[8][64][32];     // per-wave V tile, transposed: [dh][key]

  const int tid  = threadIdx.x;
  const int lane = tid & 31;
  const int wave = tid >> 5;
  const int hi = lane >> 4, l16 = lane & 15;
  const int h = blockIdx.y;
  const int qbase = blockIdx.x * 128 + wave * 16;

  const __bf16* Qh = Q  + ((size_t)h * SEQ + qbase) * HDIM;
  const __bf16* Kh = Km + (size_t)h * SEQ * HDIM;
  const __bf16* Vh = Vm + (size_t)h * SEQ * HDIM;

  // V staging assignment: 8 lanes cover one key row (8 chunks of 8 dh),
  // 4 keys per pass, 8 passes -> 32 keys. Coalesced 16B global loads.
  const int skey = lane >> 3;          // 0..3
  const int schk = (lane & 7) * 8;     // dh chunk base

  // Q fragments (contraction = head dim, split into 2 x 32)
  v16bf aq[2];
  #pragma unroll
  for (int st = 0; st < 2; ++st) {
    const __bf16* p = Qh + (size_t)l16 * HDIM + st * 32 + hi * 8;
    #pragma unroll
    for (int e = 0; e < 8; ++e) { aq[st][e] = p[e]; aq[st][e + 8] = p[16 + e]; }
  }

  float mrow[8], lrow[8];
  v8f o[4] = {};
  #pragma unroll
  for (int r = 0; r < 8; ++r) { mrow[r] = -3.0e38f; lrow[r] = 0.0f; }

  const int kend = CAUSAL ? (qbase + 16) : SEQ;
  for (int kt = 0; kt < kend; kt += 32) {
    // ---- stage V tile (32 keys x 64 dh) transposed into per-wave LDS ----
    v8bf vstage[8];
    #pragma unroll
    for (int i = 0; i < 8; ++i)
      vstage[i] = *(const v8bf*)(Vh + (size_t)(kt + i * 4 + skey) * HDIM + schk);
    #pragma unroll
    for (int i = 0; i < 8; ++i)
      #pragma unroll
      for (int j = 0; j < 8; ++j)
        Vs[wave][schk + j][i * 4 + skey] = vstage[i][j];

    // ---- scores: 16 queries x 32 keys, 2 frags x 2 K-steps ----
    v8f s[2] = {};
    #pragma unroll
    for (int nt = 0; nt < 2; ++nt)
      #pragma unroll
      for (int st = 0; st < 2; ++st) {
        v16bf bkf;
        const __bf16* p = Kh + (size_t)(kt + nt * 16 + l16) * HDIM + st * 32 + hi * 8;
        #pragma unroll
        for (int e = 0; e < 8; ++e) { bkf[e] = p[e]; bkf[e + 8] = p[16 + e]; }
        s[nt] = wmma_bf16(aq[st], bkf, s[nt]);
      }

    float sv0[8], sv1[8];
    #pragma unroll
    for (int r = 0; r < 8; ++r) {
      float a0 = s[0][r] * 0.125f;            // 1/sqrt(64)
      float a1 = s[1][r] * 0.125f;
      if (CAUSAL) {
        const int row = qbase + hi * 8 + r;
        if (kt + l16      > row) a0 = -1.0e9f;
        if (kt + 16 + l16 > row) a1 = -1.0e9f;
      }
      sv0[r] = a0; sv1[r] = a1;
    }

    // row max across the 16-lane group holding the row
    float tm[8];
    #pragma unroll
    for (int r = 0; r < 8; ++r) tm[r] = fmaxf(sv0[r], sv1[r]);
    #pragma unroll
    for (int off = 8; off > 0; off >>= 1)
      #pragma unroll
      for (int r = 0; r < 8; ++r)
        tm[r] = fmaxf(tm[r], __shfl_xor(tm[r], off, 32));

    float corr[8], ts[8];
    #pragma unroll
    for (int r = 0; r < 8; ++r) {
      const float mn = fmaxf(mrow[r], tm[r]);
      corr[r] = __expf(mrow[r] - mn);
      mrow[r] = mn;
      sv0[r] = __expf(sv0[r] - mn);
      sv1[r] = __expf(sv1[r] - mn);
      ts[r] = sv0[r] + sv1[r];
    }
    #pragma unroll
    for (int off = 8; off > 0; off >>= 1)
      #pragma unroll
      for (int r = 0; r < 8; ++r)
        ts[r] += __shfl_xor(ts[r], off, 32);
    #pragma unroll
    for (int r = 0; r < 8; ++r) lrow[r] = lrow[r] * corr[r] + ts[r];
    #pragma unroll
    for (int f = 0; f < 4; ++f)
      #pragma unroll
      for (int r = 0; r < 8; ++r) o[f][r] *= corr[r];

    // ---- stage P (16x32 bf16) through per-wave LDS ----
    #pragma unroll
    for (int r = 0; r < 8; ++r) {
      plds[wave][hi * 8 + r][l16]      = (__bf16)sv0[r];
      plds[wave][hi * 8 + r][16 + l16] = (__bf16)sv1[r];
    }
    wait_lds();   // intra-wave DS RAW (covers both plds and Vs staging)

    v16bf ap;
    {
      const __bf16* pp = &plds[wave][l16][hi * 8];
      #pragma unroll
      for (int e = 0; e < 8; ++e) { ap[e] = pp[e]; ap[e + 8] = pp[16 + e]; }
    }

    // ---- O += P @ V : B-frags from transposed LDS (contiguous reads) ----
    #pragma unroll
    for (int f = 0; f < 4; ++f) {
      v16bf bv;
      const __bf16* vp = &Vs[wave][f * 16 + l16][hi * 8];
      #pragma unroll
      for (int e = 0; e < 8; ++e) { bv[e] = vp[e]; bv[e + 8] = vp[16 + e]; }
      o[f] = wmma_bf16(ap, bv, o[f]);
    }
    wait_lds();   // reads done before next iteration overwrites plds / Vs
  }

  #pragma unroll
  for (int r = 0; r < 8; ++r) {
    const float inv = 1.0f / lrow[r];
    #pragma unroll
    for (int f = 0; f < 4; ++f)
      Out[(size_t)(qbase + hi * 8 + r) * DMODEL + h * HDIM + f * 16 + l16] =
          o[f][r] * inv;
  }
}

// ---------------------------------------------------------------------------
// y = LayerNorm(x + r) * g + b, row-wise over D=1024. One block per row.
// ---------------------------------------------------------------------------
__global__ __launch_bounds__(256)
void add_ln_kernel(const float* __restrict__ x, const float* __restrict__ r,
                   const float* __restrict__ g, const float* __restrict__ b,
                   float* __restrict__ y)
{
  __shared__ float red[256];
  const int row = blockIdx.x, t = threadIdx.x;
  const float* xr = x + (size_t)row * DMODEL;
  const float* rr = r + (size_t)row * DMODEL;

  float v[4];
  float s = 0.0f;
  #pragma unroll
  for (int i = 0; i < 4; ++i) { v[i] = xr[t + i * 256] + rr[t + i * 256]; s += v[i]; }
  red[t] = s; __syncthreads();
  for (int st = 128; st > 0; st >>= 1) { if (t < st) red[t] += red[t + st]; __syncthreads(); }
  const float mu = red[0] * (1.0f / DMODEL);
  __syncthreads();

  float var = 0.0f;
  #pragma unroll
  for (int i = 0; i < 4; ++i) { const float d = v[i] - mu; var += d * d; }
  red[t] = var; __syncthreads();
  for (int st = 128; st > 0; st >>= 1) { if (t < st) red[t] += red[t + st]; __syncthreads(); }
  const float rstd = rsqrtf(red[0] * (1.0f / DMODEL) + 1e-5f);

  #pragma unroll
  for (int i = 0; i < 4; ++i) {
    const int c = t + i * 256;
    y[(size_t)row * DMODEL + c] = (v[i] - mu) * rstd * g[c] + b[c];
  }
}

// ---------------------------------------------------------------------------
extern "C" void kernel_launch(void* const* d_in, const int* in_sizes, int n_in,
                              void* d_out, int out_size, void* d_ws, size_t ws_size,
                              hipStream_t stream)
{
  (void)in_sizes; (void)n_in; (void)out_size; (void)ws_size;
  const float* y      = (const float*)d_in[0];
  const float* enc    = (const float*)d_in[1];
  const float* Wself  = (const float*)d_in[2];
  const float* bself  = (const float*)d_in[3];
  const float* Wcross = (const float*)d_in[4];
  const float* bcross = (const float*)d_in[5];
  const float* g1 = (const float*)d_in[6];  const float* be1 = (const float*)d_in[7];
  const float* g2 = (const float*)d_in[8];  const float* be2 = (const float*)d_in[9];
  const float* g3 = (const float*)d_in[10]; const float* be3 = (const float*)d_in[11];
  const float* w1 = (const float*)d_in[12]; const float* b1  = (const float*)d_in[13];
  const float* w2 = (const float*)d_in[14]; const float* b2  = (const float*)d_in[15];
  float* out = (float*)d_out;

  // workspace carve-out (~52 MB)
  char* base = (char*)d_ws;
  size_t off = 0;
  auto take = [&](size_t bytes) -> void* {
    void* p = base + off;
    off += (bytes + 255) & ~(size_t)255;
    return p;
  };
  __bf16* qkv = (__bf16*)take((size_t)NHEAD * SEQ * HDIM * 2);  // self q=k=v
  __bf16* qc  = (__bf16*)take((size_t)NHEAD * SEQ * HDIM * 2);  // cross q
  __bf16* kvc = (__bf16*)take((size_t)NHEAD * SEQ * HDIM * 2);  // cross k=v
  float*  att = (float*) take((size_t)SEQ * DMODEL * 4);        // reused 3x
  float*  y1  = (float*) take((size_t)SEQ * DMODEL * 4);
  float*  y2  = (float*) take((size_t)SEQ * DMODEL * 4);
  __bf16* h1  = (__bf16*)take((size_t)SEQ * FFDIM * 2);

  const long sWB = (long)DMODEL * HDIM;   // per-head weight stride
  const long sWC = (long)SEQ * HDIM;      // per-head output stride

  // ---- self attention (source bug: q=k=v share one projection) ----
  gemm_ws_kernel<float, float, true, false, true>
      <<<dim3(SEQ / 64, HDIM / 64, NHEAD), 128, 0, stream>>>(
          y, Wself, bself, qkv, SEQ, HDIM, DMODEL, sWB, (long)HDIM, sWC);
  attn_kernel<true><<<dim3(SEQ / 128, NHEAD), 256, 0, stream>>>(qkv, qkv, qkv, att);
  add_ln_kernel<<<dim3(SEQ), 256, 0, stream>>>(y, att, g1, be1, y1);

  // ---- cross attention (k=v share projection, q from y1) ----
  gemm_ws_kernel<float, float, true, false, true>
      <<<dim3(SEQ / 64, HDIM / 64, NHEAD), 128, 0, stream>>>(
          y1, Wcross, bcross, qc, SEQ, HDIM, DMODEL, sWB, (long)HDIM, sWC);
  gemm_ws_kernel<float, float, true, false, true>
      <<<dim3(SEQ / 64, HDIM / 64, NHEAD), 128, 0, stream>>>(
          enc, Wcross, bcross, kvc, SEQ, HDIM, DMODEL, sWB, (long)HDIM, sWC);
  attn_kernel<false><<<dim3(SEQ / 128, NHEAD), 256, 0, stream>>>(qc, kvc, kvc, att);
  add_ln_kernel<<<dim3(SEQ), 256, 0, stream>>>(y1, att, g2, be2, y2);

  // ---- FFN ----
  gemm_ws_kernel<float, float, true, true, true>
      <<<dim3(SEQ / 64, FFDIM / 64, 1), 128, 0, stream>>>(
          y2, w1, b1, h1, SEQ, FFDIM, DMODEL, 0, 0, 0);
  gemm_ws_kernel<__bf16, float, true, false, false>
      <<<dim3(SEQ / 64, DMODEL / 64, 1), 128, 0, stream>>>(
          h1, w2, b2, att, SEQ, DMODEL, FFDIM, 0, 0, 0);
  add_ln_kernel<<<dim3(SEQ), 256, 0, stream>>>(y2, att, g3, be3, out);
}
